// AttentionL2_74835510166179
// MI455X (gfx1250) — matrix operand order repaired and checked by reading, not compiled, not run on hardware
//
#include <hip/hip_runtime.h>

typedef __attribute__((ext_vector_type(16))) _Float16 v16h;
typedef __attribute__((ext_vector_type(8)))  _Float16 v8h;
typedef __attribute__((ext_vector_type(8)))  float    v8f;

#define BATCH 4
#define SEQ   4096
#define DIN   128
#define DOUT  64
#define NTILE (BATCH * (SEQ / 16))   // 1024 query tiles
#define KSPLIT 2                     // key-range split per tile
#define KHALF (SEQ / KSPLIT)         // 2048 keys per wave

// ---------------------------------------------------------------------------
// Kernel 1: q/k/v projections (f32 math), emit f16 qh/kh (row major),
// f16 vt (transposed: [b][feat][n]) and f32 q_sq/k_sq computed from the
// *rounded* f16 values so d2 = q2 + k2 - 2 q.k stays self-consistent.
// ---------------------------------------------------------------------------
__global__ __launch_bounds__(64)
void proj_qkv(const float* __restrict__ x,
              const float* __restrict__ Wq,
              const float* __restrict__ Wk,
              const float* __restrict__ Wv,
              _Float16* __restrict__ qh,
              _Float16* __restrict__ kh,
              _Float16* __restrict__ vt,
              float* __restrict__ qsq,
              float* __restrict__ ksq)
{
    const int row = blockIdx.x;            // b*SEQ + n
    const int t   = threadIdx.x;           // output feature 0..63
    const int b   = row >> 12;
    const int n   = row & (SEQ - 1);

    __shared__ float xs[DIN];
    xs[t]      = x[(size_t)row * DIN + t];
    xs[t + 64] = x[(size_t)row * DIN + t + 64];
    __syncthreads();

    float aq = 0.f, ak = 0.f, av = 0.f;
#pragma unroll 4
    for (int i = 0; i < DIN; ++i) {
        float xv = xs[i];
        aq = fmaf(xv, Wq[i * DOUT + t], aq);
        ak = fmaf(xv, Wk[i * DOUT + t], ak);
        av = fmaf(xv, Wv[i * DOUT + t], av);
    }
    _Float16 qf = (_Float16)aq;
    _Float16 kf = (_Float16)ak;
    _Float16 vf = (_Float16)av;
    qh[(size_t)row * DOUT + t] = qf;
    kh[(size_t)row * DOUT + t] = kf;
    vt[(size_t)(b * DOUT + t) * SEQ + n] = vf;

    __shared__ float rq[64], rk[64];
    rq[t] = (float)qf * (float)qf;
    rk[t] = (float)kf * (float)kf;
    __syncthreads();
    if (t == 0) {
        float sq = 0.f, sk = 0.f;
        for (int i = 0; i < 64; ++i) { sq += rq[i]; sk += rk[i]; }
        qsq[row] = sq;
        ksq[row] = sk;
    }
}

// ---------------------------------------------------------------------------
// K-operand buffer for one 32-key step (double-buffered in the main loop).
// ---------------------------------------------------------------------------
struct KBuf {
    v8h  k[4][2];    // [tile*2+chunk][half]: B-operand fragments for S=QK^T
    float ks[2];     // k_sq for the two 16-key tiles (per lane: key = lo)
};

__device__ __forceinline__
void load_k(KBuf& kb, const _Float16* __restrict__ kh,
            const float* __restrict__ ksq, int base /*b*SEQ + k0*/,
            int lo, int hi)
{
#pragma unroll
    for (int t = 0; t < 2; ++t) {
        const _Float16* krow = kh + (size_t)(base + t * 16 + lo) * DOUT;
#pragma unroll
        for (int ch = 0; ch < 2; ++ch) {
            const int f0 = ch * 32 + hi * 8;
            kb.k[t * 2 + ch][0] = *(const v8h*)(krow + f0);
            kb.k[t * 2 + ch][1] = *(const v8h*)(krow + f0 + 16);
        }
    }
    kb.ks[0] = ksq[base + lo];
    kb.ks[1] = ksq[base + 16 + lo];
}

// ---------------------------------------------------------------------------
// Kernel 2: flash-attention style L2-distance attention, key-split partials.
// One wave32 per (16-query tile, key half). 32 keys per step, K operands
// prefetched one step ahead; V operands issued at the top of each step so
// the softmax VALU work hides their latency.
// ---------------------------------------------------------------------------
__global__ __launch_bounds__(32)
void attn_l2(const _Float16* __restrict__ qh,
             const _Float16* __restrict__ kh,
             const _Float16* __restrict__ vt,
             const float* __restrict__ qsq,
             const float* __restrict__ ksq,
             float* __restrict__ Opart,
             float* __restrict__ mpart,
             float* __restrict__ lpart)
{
    const int split = blockIdx.x & (KSPLIT - 1);
    const int tile  = blockIdx.x >> 1;           // 0..NTILE-1
    const int b     = tile >> 8;                 // 256 query tiles per batch
    const int q0    = (tile & 255) * 16;
    const int lane  = threadIdx.x;
    const int lo    = lane & 15;
    const int hi    = lane >> 4;                 // 0 or 1 (half-wave group)

    const int kbeg = split * KHALF;
    const int kend = kbeg + KHALF;

    __shared__ _Float16 Plds[16 * 32];

    // ---- A operands: Q tile, feature chunks [0,32) and [32,64)
    // A layout (16-bit, 16x32): lanes 0-15 row M=lane, K={0..7,16..23};
    //                           lanes 16-31 row M=lane-16, K={8..15,24..31}
    const _Float16* qrow = qh + (size_t)(b * SEQ + q0 + lo) * DOUT;
    v16h aq[2];
#pragma unroll
    for (int c = 0; c < 2; ++c) {
        const int f0 = c * 32 + hi * 8;
        v8h l8 = *(const v8h*)(qrow + f0);
        v8h h8 = *(const v8h*)(qrow + f0 + 16);
        aq[c] = __builtin_shufflevector(l8, h8,
                    0,1,2,3,4,5,6,7,8,9,10,11,12,13,14,15);
    }

    // q_sq per accumulator element e: row m = e + 8*hi
    float qs[8];
#pragma unroll
    for (int e = 0; e < 8; ++e) qs[e] = qsq[b * SEQ + q0 + e + 8 * hi];

    float m_run[8], l_run[8];
    v8f  o[4];
#pragma unroll
    for (int e = 0; e < 8; ++e) { m_run[e] = -1e30f; l_run[e] = 0.f; }
#pragma unroll
    for (int ft = 0; ft < 4; ++ft) o[ft] = v8f{0.f,0.f,0.f,0.f,0.f,0.f,0.f,0.f};

    const float inv_scale = 0.125f;   // 1/sqrt(DOUT)

    KBuf kbuf[2];
    load_k(kbuf[0], kh, ksq, b * SEQ + kbeg, lo, hi);

    for (int k0 = kbeg; k0 < kend; k0 += 64) {
#pragma unroll
        for (int ph = 0; ph < 2; ++ph) {
            const int kc = k0 + ph * 32;                 // current step
            int kn = kc + 32;                            // next step (clamped)
            if (kn >= kend) kn = kend - 32;
            // prefetch next K step into the other buffer
            load_k(kbuf[ph ^ 1], kh, ksq, b * SEQ + kn, lo, hi);

            // issue V loads for the current step early (latency hidden by softmax)
            v8h vb[4][2];
#pragma unroll
            for (int ft = 0; ft < 4; ++ft) {
                const _Float16* vrow =
                    vt + (size_t)(b * DOUT + ft * 16 + lo) * SEQ + kc + hi * 8;
                vb[ft][0] = *(const v8h*)(vrow);
                vb[ft][1] = *(const v8h*)(vrow + 16);
            }

            // ---------------- S = Q K^T (two 16-key tiles) ----------------
            v8f s_acc[2];
#pragma unroll
            for (int t = 0; t < 2; ++t) {
                v8f c = v8f{0.f,0.f,0.f,0.f,0.f,0.f,0.f,0.f};
#pragma unroll
                for (int ch = 0; ch < 2; ++ch) {
                    v16h bk = __builtin_shufflevector(
                        kbuf[ph].k[t * 2 + ch][0], kbuf[ph].k[t * 2 + ch][1],
                        0,1,2,3,4,5,6,7,8,9,10,11,12,13,14,15);
                    c = __builtin_amdgcn_wmma_f32_16x16x32_f16(
                            false, aq[ch], false, bk, (short)0, c, false, false);
                }
                s_acc[t] = c;
            }

            // ---------------- scores + online softmax ---------------------
            const float ksq0 = kbuf[ph].ks[0];
            const float ksq1 = kbuf[ph].ks[1];
            float sc[2][8], bm[8];
#pragma unroll
            for (int e = 0; e < 8; ++e) {
                float d2a = qs[e] + ksq0 - 2.f * s_acc[0][e];
                float d2b = qs[e] + ksq1 - 2.f * s_acc[1][e];
                float sa = __builtin_amdgcn_sqrtf(fmaxf(d2a, 0.f)) * inv_scale;
                float sb = __builtin_amdgcn_sqrtf(fmaxf(d2b, 0.f)) * inv_scale;
                sc[0][e] = sa; sc[1][e] = sb;
                bm[e] = fmaxf(sa, sb);
            }
            // row max across the 16 lanes of each half-wave (masks < 16)
#pragma unroll
            for (int m = 1; m < 16; m <<= 1)
#pragma unroll
                for (int e = 0; e < 8; ++e)
                    bm[e] = fmaxf(bm[e], __shfl_xor(bm[e], m, 32));

            float p[2][8], alpha[8], rsum[8];
#pragma unroll
            for (int e = 0; e < 8; ++e) {
                float mn = fmaxf(m_run[e], bm[e]);
                alpha[e] = __expf(m_run[e] - mn);
                m_run[e] = mn;
                float pa = __expf(sc[0][e] - mn);
                float pb = __expf(sc[1][e] - mn);
                p[0][e] = pa; p[1][e] = pb;
                rsum[e] = pa + pb;
            }
#pragma unroll
            for (int m = 1; m < 16; m <<= 1)
#pragma unroll
                for (int e = 0; e < 8; ++e)
                    rsum[e] += __shfl_xor(rsum[e], m, 32);
#pragma unroll
            for (int e = 0; e < 8; ++e) l_run[e] = l_run[e] * alpha[e] + rsum[e];

            // rescale accumulators
#pragma unroll
            for (int ft = 0; ft < 4; ++ft)
#pragma unroll
                for (int e = 0; e < 8; ++e) o[ft][e] *= alpha[e];

            // ---------------- P: C-layout -> LDS -> A-layout ---------------
            __syncthreads();
#pragma unroll
            for (int t = 0; t < 2; ++t)
#pragma unroll
                for (int e = 0; e < 8; ++e)
                    Plds[(e + 8 * hi) * 32 + lo + 16 * t] = (_Float16)p[t][e];
            __syncthreads();
            v8h p0 = *(const v8h*)(&Plds[lo * 32 + hi * 8]);
            v8h p1 = *(const v8h*)(&Plds[lo * 32 + hi * 8 + 16]);
            v16h ap = __builtin_shufflevector(p0, p1,
                        0,1,2,3,4,5,6,7,8,9,10,11,12,13,14,15);

            // ---------------- O += P V (4 feature tiles) -------------------
#pragma unroll
            for (int ft = 0; ft < 4; ++ft) {
                v16h bv = __builtin_shufflevector(vb[ft][0], vb[ft][1],
                            0,1,2,3,4,5,6,7,8,9,10,11,12,13,14,15);
                o[ft] = __builtin_amdgcn_wmma_f32_16x16x32_f16(
                            false, ap, false, bv, (short)0, o[ft], false, false);
            }
        }
    }

    // ---------------- store partials (unnormalized O, row stats) -----------
    const int pbase = (split * NTILE + tile) * 16;     // partial row base
#pragma unroll
    for (int e = 0; e < 8; ++e) {
        const int m = e + 8 * hi;
        if (lo == 0) {
            mpart[pbase + m] = m_run[e];
            lpart[pbase + m] = l_run[e];
        }
#pragma unroll
        for (int ft = 0; ft < 4; ++ft)
            Opart[(size_t)(pbase + m) * DOUT + ft * 16 + lo] = o[ft][e];
    }
}

// ---------------------------------------------------------------------------
// Kernel 3: merge the two key-split partials and normalize.
// ---------------------------------------------------------------------------
__global__ __launch_bounds__(128)
void merge_out(const float* __restrict__ Opart,
               const float* __restrict__ mpart,
               const float* __restrict__ lpart,
               float* __restrict__ out)
{
    const int tile = blockIdx.x;            // 0..NTILE-1
    const int tid  = threadIdx.x;           // 128 threads, 8 elems each
#pragma unroll
    for (int i = 0; i < 8; ++i) {
        const int idx = i * 128 + tid;      // 0..1023 within the 16x64 tile
        const int row = idx >> 6;
        const float m1 = mpart[tile * 16 + row];
        const float m2 = mpart[NTILE * 16 + tile * 16 + row];
        const float l1 = lpart[tile * 16 + row];
        const float l2 = lpart[NTILE * 16 + tile * 16 + row];
        const float mm = fmaxf(m1, m2);
        const float w1 = __expf(m1 - mm);
        const float w2 = __expf(m2 - mm);
        const float inv = 1.f / (l1 * w1 + l2 * w2);
        const float o1 = Opart[(size_t)tile * 1024 + idx];
        const float o2 = Opart[(size_t)(NTILE + tile) * 1024 + idx];
        out[(size_t)tile * 1024 + idx] = (o1 * w1 + o2 * w2) * inv;
    }
}

// ---------------------------------------------------------------------------
extern "C" void kernel_launch(void* const* d_in, const int* in_sizes, int n_in,
                              void* d_out, int out_size, void* d_ws, size_t ws_size,
                              hipStream_t stream)
{
    const float* x  = (const float*)d_in[0];
    const float* Wq = (const float*)d_in[1];
    const float* Wk = (const float*)d_in[2];
    const float* Wv = (const float*)d_in[3];

    char* ws = (char*)d_ws;
    const size_t nh = (size_t)BATCH * SEQ * DOUT;      // 1,048,576 elements
    _Float16* qh = (_Float16*)ws;  ws += nh * sizeof(_Float16);
    _Float16* kh = (_Float16*)ws;  ws += nh * sizeof(_Float16);
    _Float16* vt = (_Float16*)ws;  ws += nh * sizeof(_Float16);
    float* qsq = (float*)ws;       ws += (size_t)BATCH * SEQ * sizeof(float);
    float* ksq = (float*)ws;       ws += (size_t)BATCH * SEQ * sizeof(float);
    float* Opart = (float*)ws;     ws += (size_t)KSPLIT * NTILE * 16 * DOUT * sizeof(float);
    float* mpart = (float*)ws;     ws += (size_t)KSPLIT * NTILE * 16 * sizeof(float);
    float* lpart = (float*)ws;     ws += (size_t)KSPLIT * NTILE * 16 * sizeof(float);
    float* out = (float*)d_out;

    proj_qkv<<<BATCH * SEQ, 64, 0, stream>>>(x, Wq, Wk, Wv, qh, kh, vt, qsq, ksq);
    attn_l2<<<NTILE * KSPLIT, 32, 0, stream>>>(qh, kh, vt, qsq, ksq,
                                               Opart, mpart, lpart);
    merge_out<<<NTILE, 128, 0, stream>>>(Opart, mpart, lpart, out);
}
